// psp_net_21517786153626
// MI455X (gfx1250) — matrix operand order repaired and checked by reading, not compiled, not run on hardware
//
#include <hip/hip_runtime.h>

// ---------------------------------------------------------------------------
// CDNA5 (gfx1250) wave32 WMMA helpers
// ---------------------------------------------------------------------------
typedef __attribute__((ext_vector_type(16))) __bf16 v16bf;
typedef __attribute__((ext_vector_type(8)))  float  v8f;

union V16U { unsigned int u[8]; v16bf v; };

__device__ __forceinline__ unsigned short f2bf_u(float f) {
  unsigned int u = __builtin_bit_cast(unsigned int, f);
  u += 0x7FFFu + ((u >> 16) & 1u);          // round-to-nearest-even
  return (unsigned short)(u >> 16);
}
__device__ __forceinline__ float bf2f(unsigned short h) {
  return __builtin_bit_cast(float, ((unsigned int)h) << 16);
}
__device__ __forceinline__ float sigm(float x) { return 1.f / (1.f + expf(-x)); }

// A-matrix fragment (16x32 bf16): lane L holds row M=L%16,
// K=kbase+(L/16)*16..+15 contiguous (matches load_b_frag).
__device__ __forceinline__ v16bf load_a_frag(const unsigned short* s, int row0,
                                             int ldk, int kbase, int lane) {
  int row = row0 + (lane & 15);
  const unsigned int* p = reinterpret_cast<const unsigned int*>(
      s + row * ldk + kbase + ((lane >> 4) << 4));
  V16U t;
#pragma unroll
  for (int r = 0; r < 4; ++r) { t.u[r] = p[r]; t.u[r + 4] = p[r + 4]; }
  return t.v;
}

// B-matrix fragment (32x16 bf16): lane L holds column N=L%16; lanes 0-15 carry
// K=kbase..+15, lanes 16-31 carry K=kbase+16..+31 (contiguous 16 shorts).
// W is row-major [N, ldk] so B[k][n] = W[n][k]  (i.e. computes A @ W^T).
__device__ __forceinline__ v16bf load_b_frag(const unsigned short* w, int ldk,
                                             int n0, int kbase, int lane) {
  int n = n0 + (lane & 15);
  const uint4* q = reinterpret_cast<const uint4*>(
      w + n * ldk + kbase + ((lane >> 4) << 4));
  uint4 x0 = q[0], x1 = q[1];
  V16U t;
  t.u[0] = x0.x; t.u[1] = x0.y; t.u[2] = x0.z; t.u[3] = x0.w;
  t.u[4] = x1.x; t.u[5] = x1.y; t.u[6] = x1.z; t.u[7] = x1.w;
  return t.v;
}

__device__ __forceinline__ v8f wmma_bf16(v16bf a, v16bf b, v8f c) {
  return __builtin_amdgcn_wmma_f32_16x16x32_bf16(false, a, false, b, (short)0,
                                                 c, false, false);
}

// ---------------------------------------------------------------------------
// Init kernels: fp32 -> bf16 weight conversion, LSTM state zeroing
// ---------------------------------------------------------------------------
__global__ void convert_bf16_kernel(const float* __restrict__ src,
                                    unsigned short* __restrict__ dst, int n) {
  int i = blockIdx.x * 256 + threadIdx.x;
  if (i < n) dst[i] = f2bf_u(src[i]);
}

__global__ void convert_pad64_kernel(const float* __restrict__ src,
                                     unsigned short* __restrict__ dst,
                                     int rows) {
  int i = blockIdx.x * 256 + threadIdx.x;
  if (i < 64 * 512) {
    int r = i >> 9;
    dst[i] = (r < rows) ? f2bf_u(src[i]) : (unsigned short)0;
  }
}

__global__ void zero_state_kernel(unsigned short* hb, float* c, int n) {
  int i = blockIdx.x * 256 + threadIdx.x;
  if (i < n) { hb[i] = 0; c[i] = 0.f; }
}

// ---------------------------------------------------------------------------
// Fused AVGA spatial-attention kernel: one workgroup per (b,t) sample.
//   vt = relu(V[49,512] @ vid_w^T + vid_b)      (WMMA, 64x512x512)
//   content = vt @ av_w^T (+ at@ag_w^T bcast)   (WMMA, 64x64x512)
//   z -> softmax(49) -> pooled V -> 512->128 projection
// Register blocking: each wave owns 4 N-tiles and iterates all 4 M-tiles with
// 4 live accumulators (32 acc VGPRs) -> no loop-invariant fragment hoisting,
// no scratch spills, and vid_w is read exactly once per block.
// ---------------------------------------------------------------------------
__global__ void avga_kernel(const float* __restrict__ audio,
                            const float* __restrict__ video,
                            const unsigned short* __restrict__ vid_w_bf,
                            const float* __restrict__ vid_b,
                            const float* __restrict__ aud_w,
                            const float* __restrict__ aud_b,
                            const unsigned short* __restrict__ av_w_bf,
                            const float* __restrict__ ag_w,
                            const float* __restrict__ ah_w,
                            const float* __restrict__ lv_w,
                            const float* __restrict__ lv_b,
                            float* __restrict__ video_t) {
  extern __shared__ char smem1[];
  unsigned short* sV  = (unsigned short*)smem1;   // 64*512 bf16   (65536 B)
  unsigned short* sVT = sV + 64 * 512;            // 64*512 bf16   (65536 B)
  float* sC   = (float*)(sVT + 64 * 512);         // 64*64 f32
  float* sAud = sC + 64 * 64;                     // 128
  float* sAt  = sAud + 128;                       // 512
  float* sAtg = sAt + 512;                        // 64
  float* sZ   = sAtg + 64;                        // 64
  float* sAl  = sZ + 64;                          // 64
  float* sVt  = sAl + 64;                         // 512

  const int n   = blockIdx.x;                     // 0..2559 = b*10+t
  const int tid = threadIdx.x;
  const int lane = tid & 31, wv = tid >> 5;

  if (tid < 128) sAud[tid] = audio[(size_t)n * 128 + tid];
  const float* Vn = video + (size_t)n * 49 * 512;
  for (int idx = tid; idx < 64 * 512; idx += 256) {
    int s = idx >> 9;
    sV[idx] = (s < 49) ? f2bf_u(Vn[idx]) : (unsigned short)0;
  }
  __syncthreads();

  // at = relu(audio_n @ aud_w^T + aud_b)
  for (int j = tid; j < 512; j += 256) {
    float acc = aud_b[j];
    const float* wr = aud_w + (size_t)j * 128;
    for (int k = 0; k < 128; ++k) acc = fmaf(sAud[k], wr[k], acc);
    sAt[j] = fmaxf(acc, 0.f);
  }
  __syncthreads();

  // atg[j] = at @ ag_w[j]
  if (tid < 64) {
    float acc = 0.f;
    if (tid < 49) {
      const float* wr = ag_w + (size_t)tid * 512;
      for (int k = 0; k < 512; ++k) acc = fmaf(sAt[k], wr[k], acc);
    }
    sAtg[tid] = acc;
  }

  // GEMM1: vt = relu(sV @ vid_w^T + vid_b) -> sVT (bf16).
  // Wave w owns ntiles w*4 .. w*4+3; for each, 4 M-tile accumulators.
#pragma unroll 1
  for (int jn = 0; jn < 4; ++jn) {
    const int ntile = wv * 4 + jn;
    v8f acc[4] = {};
#pragma unroll 1
    for (int ks = 0; ks < 16; ++ks) {
      v16bf b = load_b_frag(vid_w_bf, 512, ntile * 16, ks * 32, lane);
#pragma unroll
      for (int mt = 0; mt < 4; ++mt) {
        v16bf a = load_a_frag(sV, mt * 16, 512, ks * 32, lane);
        acc[mt] = wmma_bf16(a, b, acc[mt]);
      }
    }
    const int col = ntile * 16 + (lane & 15);
    const float bias = vid_b[col];
#pragma unroll
    for (int mt = 0; mt < 4; ++mt) {
#pragma unroll
      for (int i = 0; i < 8; ++i) {
        int row = mt * 16 + i + ((lane >> 4) << 3);
        sVT[row * 512 + col] = f2bf_u(fmaxf(acc[mt][i] + bias, 0.f));
      }
    }
  }
  __syncthreads();

  // GEMM2: content = sVT @ av_w^T  [64x64]; ks-outer, 2 live accumulators
  {
    const int mtile = wv & 3, nt0 = (wv >> 2) * 2;
    v8f acc0 = {}, acc1 = {};
#pragma unroll 1
    for (int ks = 0; ks < 16; ++ks) {
      v16bf a  = load_a_frag(sVT, mtile * 16, 512, ks * 32, lane);
      v16bf b0 = load_b_frag(av_w_bf, 512, nt0 * 16, ks * 32, lane);
      v16bf b1 = load_b_frag(av_w_bf, 512, nt0 * 16 + 16, ks * 32, lane);
      acc0 = wmma_bf16(a, b0, acc0);
      acc1 = wmma_bf16(a, b1, acc1);
    }
#pragma unroll
    for (int t = 0; t < 2; ++t) {
      v8f& acc = t ? acc1 : acc0;
      int col = (nt0 + t) * 16 + (lane & 15);
#pragma unroll
      for (int i = 0; i < 8; ++i) {
        int row = mtile * 16 + i + ((lane >> 4) << 3);
        sC[row * 64 + col] = acc[i];
      }
    }
  }
  __syncthreads();

  // z[s] = sum_j tanh(content[s,j] + atg[j]) * ah_w[j]; softmax over 49
  if (tid < 49) {
    float acc = 0.f;
    for (int j = 0; j < 49; ++j)
      acc = fmaf(tanhf(sC[tid * 64 + j] + sAtg[j]), ah_w[j], acc);
    sZ[tid] = acc;
  }
  __syncthreads();
  if (tid == 0) {
    float m = sZ[0];
    for (int s = 1; s < 49; ++s) m = fmaxf(m, sZ[s]);
    float sum = 0.f;
    for (int s = 0; s < 49; ++s) { float e = expf(sZ[s] - m); sAl[s] = e; sum += e; }
    float inv = 1.f / sum;
    for (int s = 0; s < 49; ++s) sAl[s] *= inv;
  }
  __syncthreads();

  // attention-pooled V (512), then project to 128 via lv_w
  for (int d = tid; d < 512; d += 256) {
    float acc = 0.f;
    for (int s = 0; s < 49; ++s) acc = fmaf(sAl[s], bf2f(sV[s * 512 + d]), acc);
    sVt[d] = acc;
  }
  __syncthreads();
  if (tid < 128) {
    float acc = lv_b[tid];
    const float* wr = lv_w + (size_t)tid * 512;
    for (int d = 0; d < 512; ++d) acc = fmaf(sVt[d], wr[d], acc);
    video_t[(size_t)n * 128 + tid] = acc;
  }
}

// ---------------------------------------------------------------------------
// Generic bf16 WMMA GEMM: C[M,N] = act(A[M,K] @ B[N,K]^T + bias1 + bias2)
// 64x64 C tile per 256-thread block; M,N multiples of 64; K multiple of 64.
// K-step 64 per LDS stage (4 WMMAs between barriers).
// ---------------------------------------------------------------------------
__global__ void gemm_bf16_kernel(const float* __restrict__ A,
                                 const float* __restrict__ B,
                                 const float* bias1, const float* bias2,
                                 float* __restrict__ C, int M, int N, int K,
                                 int relu_flag) {
  __shared__ __align__(16) unsigned short sA[64 * 64];
  __shared__ __align__(16) unsigned short sB[64 * 64];
  const int tid = threadIdx.x, lane = tid & 31, wv = tid >> 5;
  const int m0 = blockIdx.x * 64, n0 = blockIdx.y * 64;
  const int mtile = wv & 3, nt0 = (wv >> 2) * 2;
  v8f acc0 = {}, acc1 = {};
  for (int k0 = 0; k0 < K; k0 += 64) {
    __syncthreads();
    for (int idx = tid; idx < 64 * 64; idx += 256) {
      int r = idx >> 6, c = idx & 63;
      sA[idx] = f2bf_u(A[(size_t)(m0 + r) * K + k0 + c]);
      sB[idx] = f2bf_u(B[(size_t)(n0 + r) * K + k0 + c]);
    }
    __syncthreads();
#pragma unroll
    for (int ks = 0; ks < 2; ++ks) {
      v16bf a  = load_a_frag(sA, mtile * 16, 64, ks * 32, lane);
      v16bf b0 = load_b_frag(sB, 64, nt0 * 16, ks * 32, lane);
      v16bf b1 = load_b_frag(sB, 64, nt0 * 16 + 16, ks * 32, lane);
      acc0 = wmma_bf16(a, b0, acc0);
      acc1 = wmma_bf16(a, b1, acc1);
    }
  }
#pragma unroll
  for (int t = 0; t < 2; ++t) {
    v8f& acc = t ? acc1 : acc0;
    int colN = n0 + (nt0 + t) * 16 + (lane & 15);
    float bs = (bias1 ? bias1[colN] : 0.f) + (bias2 ? bias2[colN] : 0.f);
#pragma unroll
    for (int i = 0; i < 8; ++i) {
      int row = m0 + mtile * 16 + i + ((lane >> 4) << 3);
      float v = acc[i] + bs;
      if (relu_flag) v = fmaxf(v, 0.f);
      C[(size_t)row * N + colN] = v;
    }
  }
}

// ---------------------------------------------------------------------------
// One LSTM timestep for one direction: g = xg_t + h @ whh^T (WMMA), then
// gate nonlinearity + c/h update.  B=256 (4 blocks of 64 rows), H=128, T=10.
// Same 4-ntile x 4-mtile register blocking as AVGA GEMM1.
// ---------------------------------------------------------------------------
__global__ void lstm_step_kernel(const float* __restrict__ xg,
                                 const unsigned short* __restrict__ whh_bf,
                                 unsigned short* __restrict__ h_bf,
                                 float* __restrict__ c_st,
                                 float* __restrict__ h_out, int t_actual,
                                 int col_off) {
  extern __shared__ char smem3[];
  unsigned short* sH = (unsigned short*)smem3;  // 64*128 bf16 (16384 B)
  float* sG = (float*)(sH + 64 * 128);          // 64*512 f32  (131072 B)
  const int tid = threadIdx.x, lane = tid & 31, wv = tid >> 5;
  const int b0 = blockIdx.x * 64;

  for (int idx = tid; idx < 64 * 128; idx += 256)
    sH[idx] = h_bf[(size_t)b0 * 128 + idx];
  __syncthreads();

#pragma unroll 1
  for (int jn = 0; jn < 4; ++jn) {
    const int ntile = wv * 4 + jn;
    v8f acc[4] = {};
#pragma unroll
    for (int ks = 0; ks < 4; ++ks) {
      v16bf b = load_b_frag(whh_bf, 128, ntile * 16, ks * 32, lane);
#pragma unroll
      for (int mt = 0; mt < 4; ++mt) {
        v16bf a = load_a_frag(sH, mt * 16, 128, ks * 32, lane);
        acc[mt] = wmma_bf16(a, b, acc[mt]);
      }
    }
    const int col = ntile * 16 + (lane & 15);
#pragma unroll
    for (int mt = 0; mt < 4; ++mt) {
#pragma unroll
      for (int i = 0; i < 8; ++i) {
        int rl = mt * 16 + i + ((lane >> 4) << 3);
        int bg = b0 + rl;
        sG[rl * 512 + col] =
            acc[mt][i] + xg[((size_t)bg * 10 + t_actual) * 512 + col];
      }
    }
  }
  __syncthreads();

  for (int idx = tid; idx < 64 * 128; idx += 256) {
    int rl = idx >> 7, j = idx & 127;
    int bg = b0 + rl;
    float i_ = sigm(sG[rl * 512 + j]);
    float f_ = sigm(sG[rl * 512 + 128 + j]);
    float g_ = tanhf(sG[rl * 512 + 256 + j]);
    float o_ = sigm(sG[rl * 512 + 384 + j]);
    float c = f_ * c_st[(size_t)bg * 128 + j] + i_ * g_;
    float h = o_ * tanhf(c);
    c_st[(size_t)bg * 128 + j] = c;
    h_bf[(size_t)bg * 128 + j] = f2bf_u(h);
    h_out[((size_t)bg * 10 + t_actual) * 256 + col_off + j] = h;
  }
}

// ---------------------------------------------------------------------------
// Fused PSP block, one workgroup per batch element (T=10, D=256):
// raw cross-attention, normalize+threshold, cross features, residual + LN,
// fuse = 0.5*(fv + fa)
// ---------------------------------------------------------------------------
__global__ void psp_kernel(const float* __restrict__ la,
                           const float* __restrict__ lv,
                           const float* __restrict__ v1g,
                           const float* __restrict__ v2g,
                           const float* __restrict__ a1g,
                           const float* __restrict__ a2g,
                           const float* __restrict__ thr_p,
                           const float* __restrict__ ln_g,
                           const float* __restrict__ ln_b,
                           float* __restrict__ fuse) {
  extern __shared__ char smem4[];
  float* sLa = (float*)smem4;        // each 10*256
  float* sLv = sLa + 2560;
  float* sV1 = sLv + 2560;
  float* sV2 = sV1 + 2560;
  float* sA1 = sV2 + 2560;
  float* sA2 = sA1 + 2560;
  float* sCv = sA2 + 2560;
  float* sCa = sCv + 2560;
  float* sRaw = sCa + 2560;          // 100
  float* sVa = sRaw + 100;           // 100
  float* sAv = sVa + 100;            // 100

  const int b = blockIdx.x, tid = threadIdx.x;
  const size_t base = (size_t)b * 2560;
  for (int i = tid; i < 2560; i += 256) {
    sLa[i] = la[base + i]; sLv[i] = lv[base + i];
    sV1[i] = v1g[base + i]; sV2[i] = v2g[base + i];
    sA1[i] = a1g[base + i]; sA2[i] = a2g[base + i];
  }
  __syncthreads();

  if (tid < 100) {
    int t = tid / 10, s = tid % 10;
    float acc = 0.f;
    for (int d = 0; d < 256; ++d) acc = fmaf(sV2[t * 256 + d], sA1[s * 256 + d], acc);
    sRaw[tid] = fmaxf(acc * 0.0625f, 0.f);   // / sqrt(256)
  }
  __syncthreads();

  const float thr = thr_p[0];
  if (tid < 10) {
    float sum = 0.f;
    for (int s = 0; s < 10; ++s) sum += sRaw[tid * 10 + s];
    float inv = 1.f / (sum + 1e-8f);
    for (int s = 0; s < 10; ++s) {
      float v = sRaw[tid * 10 + s] * inv;
      sVa[tid * 10 + s] = (v > thr) ? v : 0.f;
    }
    float sum2 = 0.f;
    for (int s = 0; s < 10; ++s) sum2 += sRaw[s * 10 + tid];
    float inv2 = 1.f / (sum2 + 1e-8f);
    for (int s = 0; s < 10; ++s) {
      float v = sRaw[s * 10 + tid] * inv2;
      sAv[tid * 10 + s] = (v > thr) ? v : 0.f;
    }
  }
  __syncthreads();

  for (int idx = tid; idx < 2560; idx += 256) {
    int t = idx >> 8, d = idx & 255;
    float cv = 0.f, ca = 0.f;
    for (int s = 0; s < 10; ++s) {
      cv = fmaf(sVa[t * 10 + s], sA2[s * 256 + d], cv);
      ca = fmaf(sAv[t * 10 + s], sV1[s * 256 + d], ca);
    }
    sCv[idx] = sLv[idx] + cv;   // pre-LN fv
    sCa[idx] = sLa[idx] + ca;   // pre-LN fa
  }
  __syncthreads();

  // LayerNorm(256) per row via wave32 shuffle reductions; 20 rows / 8 waves
  const int lane = tid & 31, wv = tid >> 5;
  for (int r = wv; r < 20; r += 8) {
    float* x = (r < 10) ? (sCv + r * 256) : (sCa + (r - 10) * 256);
    float s1 = 0.f;
    for (int d = lane; d < 256; d += 32) s1 += x[d];
    for (int o = 16; o > 0; o >>= 1) s1 += __shfl_xor(s1, o);
    float mu = s1 * (1.f / 256.f);
    float s2 = 0.f;
    for (int d = lane; d < 256; d += 32) { float df = x[d] - mu; s2 += df * df; }
    for (int o = 16; o > 0; o >>= 1) s2 += __shfl_xor(s2, o);
    float rstd = rsqrtf(s2 * (1.f / 256.f) + 1e-6f);
    for (int d = lane; d < 256; d += 32)
      x[d] = (x[d] - mu) * rstd * ln_g[d] + ln_b[d];
  }
  __syncthreads();
  for (int idx = tid; idx < 2560; idx += 256)
    fuse[base + idx] = 0.5f * (sCv[idx] + sCa[idx]);
}

// ---------------------------------------------------------------------------
// Head: score = hcat @ l2_w^T, temporal gate, write score + softmax(mean_t)
// ---------------------------------------------------------------------------
__global__ void head_kernel(const float* __restrict__ hcat,
                            const float* __restrict__ l2_w,
                            const float* __restrict__ w3_w,
                            float* __restrict__ out) {
  __shared__ float sH[10 * 64];
  __shared__ float sS[10 * 32];
  __shared__ float sM[32];
  __shared__ float sInv[1];
  const int b = blockIdx.x, tid = threadIdx.x;
  for (int i = tid; i < 640; i += 256) sH[i] = hcat[(size_t)b * 640 + i];
  __syncthreads();
  for (int i = tid; i < 290; i += 256) {
    int t = i / 29, c = i % 29;
    float acc = 0.f;
    for (int k = 0; k < 64; ++k) acc = fmaf(sH[t * 64 + k], l2_w[c * 64 + k], acc);
    sS[t * 32 + c] = acc;
  }
  __syncthreads();
  if (tid < 10) {
    float acc = 0.f;
    for (int c = 0; c < 29; ++c) acc = fmaf(sS[tid * 32 + c], w3_w[c], acc);
    float tw = sigm(fmaxf(acc, 0.f));
    for (int c = 0; c < 29; ++c) sS[tid * 32 + c] *= tw;
  }
  __syncthreads();
  for (int i = tid; i < 290; i += 256) {
    int t = i / 29, c = i % 29;
    out[7424 + ((size_t)b * 10 + t) * 29 + c] = sS[t * 32 + c];
  }
  if (tid < 29) {
    float m = 0.f;
    for (int t = 0; t < 10; ++t) m += sS[t * 32 + tid];
    sM[tid] = m * 0.1f;
  }
  __syncthreads();
  if (tid == 0) {
    float mx = sM[0];
    for (int c = 1; c < 29; ++c) mx = fmaxf(mx, sM[c]);
    float sum = 0.f;
    for (int c = 0; c < 29; ++c) { float e = expf(sM[c] - mx); sM[c] = e; sum += e; }
    sInv[0] = 1.f / sum;
  }
  __syncthreads();
  if (tid < 29) out[(size_t)b * 29 + tid] = sM[tid] * sInv[0];
}

// ---------------------------------------------------------------------------
// Host launcher
// ---------------------------------------------------------------------------
extern "C" void kernel_launch(void* const* d_in, const int* in_sizes, int n_in,
                              void* d_out, int out_size, void* d_ws,
                              size_t ws_size, hipStream_t stream) {
  const float* audio = (const float*)d_in[0];   // [256,10,128]
  const float* video = (const float*)d_in[1];   // [256,10,7,7,512]
  const float* thr   = (const float*)d_in[2];
  const float* aud_w = (const float*)d_in[3];
  const float* aud_b = (const float*)d_in[4];
  const float* vid_w = (const float*)d_in[5];
  const float* vid_b = (const float*)d_in[6];
  const float* av_w  = (const float*)d_in[7];
  const float* ag_w  = (const float*)d_in[8];
  const float* ah_w  = (const float*)d_in[9];
  const float* lv_w  = (const float*)d_in[10];
  const float* lv_b  = (const float*)d_in[11];
  // LSTM params: dirs 0=la_f, 1=la_r, 2=lvd_f, 3=lvd_r
  const float* wih[4]; const float* whh[4]; const float* bih[4]; const float* bhh[4];
  for (int d = 0; d < 4; ++d) {
    wih[d] = (const float*)d_in[12 + 4 * d + 0];
    whh[d] = (const float*)d_in[12 + 4 * d + 1];
    bih[d] = (const float*)d_in[12 + 4 * d + 2];
    bhh[d] = (const float*)d_in[12 + 4 * d + 3];
  }
  const float* p_vl1 = (const float*)d_in[28];
  const float* p_vl2 = (const float*)d_in[29];
  const float* p_al1 = (const float*)d_in[30];
  const float* p_al2 = (const float*)d_in[31];
  const float* ln_g  = (const float*)d_in[32];
  const float* ln_b  = (const float*)d_in[33];
  const float* l1_w  = (const float*)d_in[34];
  const float* l2_w  = (const float*)d_in[35];
  const float* w3_w  = (const float*)d_in[36];
  float* out = (float*)d_out;

  // workspace carve-out (≈43 MB)
  char* ws = (char*)d_ws;
  size_t off = 0;
  auto carve = [&](size_t bytes) {
    void* p = ws + off;
    off += (bytes + 255) & ~size_t(255);
    return p;
  };
  unsigned short* vidw_bf = (unsigned short*)carve(512 * 512 * 2);
  unsigned short* avw_bf  = (unsigned short*)carve(64 * 512 * 2);
  unsigned short* whh_bf  = (unsigned short*)carve(4 * 512 * 128 * 2);
  float* video_t = (float*)carve((size_t)2560 * 128 * 4);
  float* xg      = (float*)carve((size_t)4 * 2560 * 512 * 4);
  unsigned short* h_bf = (unsigned short*)carve(4 * 256 * 128 * 2);
  float* c_st    = (float*)carve(4 * 256 * 128 * 4);
  float* la_buf  = (float*)carve((size_t)2560 * 256 * 4);
  float* lv_buf  = (float*)carve((size_t)2560 * 256 * 4);
  float* v1b     = (float*)carve((size_t)2560 * 256 * 4);
  float* v2b     = (float*)carve((size_t)2560 * 256 * 4);
  float* a1b     = (float*)carve((size_t)2560 * 256 * 4);
  float* a2b     = (float*)carve((size_t)2560 * 256 * 4);
  float* fuse    = (float*)carve((size_t)2560 * 256 * 4);
  float* hcat    = (float*)carve((size_t)2560 * 64 * 4);

  // --- init: weight conversion + recurrent state zeroing ---
  convert_bf16_kernel<<<(512 * 512 + 255) / 256, 256, 0, stream>>>(vid_w, vidw_bf, 512 * 512);
  convert_pad64_kernel<<<128, 256, 0, stream>>>(av_w, avw_bf, 49);
  for (int d = 0; d < 4; ++d)
    convert_bf16_kernel<<<256, 256, 0, stream>>>(whh[d], whh_bf + d * 65536, 65536);
  zero_state_kernel<<<512, 256, 0, stream>>>(h_bf, c_st, 4 * 256 * 128);

  // --- AVGA (dominant GEMM, WMMA) ---
  const size_t K1_LDS = 2 * 64 * 512 * 2 +
                        (64 * 64 + 128 + 512 + 64 + 64 + 64 + 512) * 4;
  avga_kernel<<<2560, 256, K1_LDS, stream>>>(audio, video, vidw_bf, vid_b,
                                             aud_w, aud_b, avw_bf, ag_w, ah_w,
                                             lv_w, lv_b, video_t);

  // --- LSTM input-gate precompute: xg = x @ wih^T + bih + bhh ---
  dim3 gXG(40, 8);  // M=2560, N=512
  gemm_bf16_kernel<<<gXG, 256, 0, stream>>>(audio,   wih[0], bih[0], bhh[0], xg + 0 * (size_t)2560 * 512, 2560, 512, 128, 0);
  gemm_bf16_kernel<<<gXG, 256, 0, stream>>>(audio,   wih[1], bih[1], bhh[1], xg + 1 * (size_t)2560 * 512, 2560, 512, 128, 0);
  gemm_bf16_kernel<<<gXG, 256, 0, stream>>>(video_t, wih[2], bih[2], bhh[2], xg + 2 * (size_t)2560 * 512, 2560, 512, 128, 0);
  gemm_bf16_kernel<<<gXG, 256, 0, stream>>>(video_t, wih[3], bih[3], bhh[3], xg + 3 * (size_t)2560 * 512, 2560, 512, 128, 0);

  // --- recurrent steps (forward dirs 0,2; reverse dirs 1,3) ---
  const size_t K3_LDS = 64 * 128 * 2 + 64 * 512 * 4;
  for (int d = 0; d < 4; ++d) {
    float* hout = (d < 2) ? la_buf : lv_buf;
    int col_off = (d & 1) ? 128 : 0;
    for (int s = 0; s < 10; ++s) {
      int t_actual = (d & 1) ? (9 - s) : s;
      lstm_step_kernel<<<4, 256, K3_LDS, stream>>>(
          xg + (size_t)d * 2560 * 512, whh_bf + d * 65536,
          h_bf + (size_t)d * 256 * 128, c_st + (size_t)d * 256 * 128, hout,
          t_actual, col_off);
    }
  }

  // --- PSP feedforward layers (relu, no bias) ---
  dim3 gFF(40, 4);  // M=2560, N=256, K=256
  gemm_bf16_kernel<<<gFF, 256, 0, stream>>>(lv_buf, p_vl1, nullptr, nullptr, v1b, 2560, 256, 256, 1);
  gemm_bf16_kernel<<<gFF, 256, 0, stream>>>(lv_buf, p_vl2, nullptr, nullptr, v2b, 2560, 256, 256, 1);
  gemm_bf16_kernel<<<gFF, 256, 0, stream>>>(la_buf, p_al1, nullptr, nullptr, a1b, 2560, 256, 256, 1);
  gemm_bf16_kernel<<<gFF, 256, 0, stream>>>(la_buf, p_al2, nullptr, nullptr, a2b, 2560, 256, 256, 1);

  // --- fused PSP cross-attention + LN + fuse ---
  const size_t K4_LDS = (2560 * 8 + 304) * 4;
  psp_kernel<<<256, 256, K4_LDS, stream>>>(la_buf, lv_buf, v1b, v2b, a1b, a2b,
                                           thr, ln_g, ln_b, fuse);

  // --- classifier ---
  dim3 gL1(40, 1);  // M=2560, N=64, K=256
  gemm_bf16_kernel<<<gL1, 256, 0, stream>>>(fuse, l1_w, nullptr, nullptr, hcat, 2560, 64, 256, 1);
  head_kernel<<<256, 256, 0, stream>>>(hcat, l2_w, w3_w, out);
}
